// OpenGrammarModel_62474594287803
// MI455X (gfx1250) — matrix operand order repaired and checked by reading, not compiled, not run on hardware
//
#include <hip/hip_runtime.h>
#include <hip/hip_bf16.h>

typedef __attribute__((ext_vector_type(16))) _Float16 v16h;
typedef __attribute__((ext_vector_type(8)))  _Float16 v8h;
typedef __attribute__((ext_vector_type(8)))  float    v8f;

#define BB 4
#define SS 2048
#define HH 1024
#define LL 4
#define NN 4
#define VV 256
#define DD 256
#define BS (BB*SS)          // 8192
#define SH (SS*HH)          // 2^21
#define EW ((BS*HH)/256)    // 32768 elementwise blocks

// ---------------------------------------------------------------------------
// WMMA f16 GEMM:  C[M x Nout] = A[M x K] * W[Nout x K]^T (+bias)(+=C)
// Block = 256 threads = 8 waves as 4(M) x 2(N); wave tile = 32M x 64N.
// B k-slices async-staged to LDS (double buffered); A double buffered in regs.
// ---------------------------------------------------------------------------
template<bool ACC, bool HASB>
__global__ __launch_bounds__(256)
void wgemm_kernel(const _Float16* __restrict__ A,
                  const _Float16* __restrict__ W,
                  const float* __restrict__ bias,
                  float* __restrict__ C,
                  int M, int K, int Nout)
{
    __shared__ alignas(64) _Float16 bt[2][128 * 32];

    const int tid  = threadIdx.x;
    const int lane = tid & 31;
    const int wave = tid >> 5;
    const int lm = lane & 15;
    const int lh = lane >> 4;
    const int wm = wave >> 1;        // 0..3  (M)
    const int wn = wave & 1;         // 0..1  (N)
    const int mb  = blockIdx.y * 128 + wm * 32;
    const int nb0 = blockIdx.x * 128;
    const int NK  = K >> 5;

    const int srow = tid >> 1;
    const int sho  = (tid & 1) * 16;
    const _Float16* sbase = W + (size_t)(nb0 + srow) * K + sho;
    const unsigned lb0 = (unsigned)(size_t)&bt[0][srow * 32 + sho];
    const unsigned lb1 = (unsigned)(size_t)&bt[1][srow * 32 + sho];

    auto STAGE = [&](int bufsel, int kb) {
        unsigned la = bufsel ? lb1 : lb0;
        unsigned long long ga = (unsigned long long)(size_t)(sbase + kb);
        asm volatile("global_load_async_to_lds_b128 %0, %1, off"
                     :: "v"(la), "v"(ga) : "memory");
        asm volatile("global_load_async_to_lds_b128 %0, %1, off offset:16"
                     :: "v"(la), "v"(ga) : "memory");
    };

    const v8h* arow0 = (const v8h*)(A + (size_t)(mb + lm) * K);
    const v8h* arow1 = (const v8h*)(A + (size_t)(mb + 16 + lm) * K);
    v16h a[2][2];
    auto LOADA = [&](int set, int kb) {
        const int k8 = kb >> 3;
        v8h x0 = arow0[k8 + lh], x1 = arow0[k8 + 2 + lh];
        v8h y0 = arow1[k8 + lh], y1 = arow1[k8 + 2 + lh];
        #pragma unroll
        for (int i = 0; i < 8; ++i) {
            a[set][0][i] = x0[i]; a[set][0][8 + i] = x1[i];
            a[set][1][i] = y0[i]; a[set][1][8 + i] = y1[i];
        }
    };

    STAGE(0, 0);
    STAGE(1, 32);
    LOADA(0, 0);
    LOADA(1, 32);

    v8f c[2][4] = {};

    for (int k = 0; k < NK; ++k) {
        const int cur = k & 1;
        if (k + 1 < NK) asm volatile("s_wait_asynccnt 0x2" ::: "memory");
        else            asm volatile("s_wait_asynccnt 0x0" ::: "memory");
        __syncthreads();

        v16h b[4];
        #pragma unroll
        for (int f = 0; f < 4; ++f) {
            const v8h* p = (const v8h*)&bt[cur][(wn * 64 + f * 16 + lm) * 32 + lh * 8];
            v8h t0 = p[0], t1 = p[2];
            #pragma unroll
            for (int i = 0; i < 8; ++i) { b[f][i] = t0[i]; b[f][8 + i] = t1[i]; }
        }
        asm volatile("s_wait_dscnt 0x0" ::: "memory");
        __syncthreads();
        if (k + 2 < NK) STAGE(cur, (k + 2) * 32);

        #pragma unroll
        for (int f = 0; f < 4; ++f)
            c[0][f] = __builtin_amdgcn_wmma_f32_16x16x32_f16(false, a[cur][0], false, b[f], (short)0, c[0][f], false, false);
        #pragma unroll
        for (int f = 0; f < 4; ++f)
            c[1][f] = __builtin_amdgcn_wmma_f32_16x16x32_f16(false, a[cur][1], false, b[f], (short)0, c[1][f], false, false);

        if (k + 2 < NK) LOADA(cur, (k + 2) * 32);
    }

    #pragma unroll
    for (int g = 0; g < 2; ++g) {
        const int rowb = mb + g * 16 + lh * 8;
        #pragma unroll
        for (int f = 0; f < 4; ++f) {
            const int n = nb0 + wn * 64 + f * 16 + lm;
            const float bv = HASB ? bias[n] : 0.0f;
            #pragma unroll
            for (int r = 0; r < 8; ++r) {
                size_t idx = (size_t)(rowb + r) * Nout + n;
                float val = c[g][f][r] + bv;
                if (ACC) val += C[idx];
                C[idx] = val;
            }
        }
    }
}

// ---------------------------------------------------------------------------
// Flash attention with async-staged K/V tiles shared by all 4 waves.
// Block = 128 threads = 4 waves, each wave 16 query rows x full D=256.
// Per 32-key step: K tile (32x256) and Vt tile (256x32) are staged to LDS via
// global_load_async_to_lds_b128 (double buffered, ASYNCcnt handoff); score and
// PV fragments are ds_load reads; P re-fragments through a per-wave LDS tile.
// Q,K layout: (B,S,H) f16 head slice n*D.  Vt layout: (B*N, D, S) f16.
// ---------------------------------------------------------------------------
__global__ __launch_bounds__(128)
void attn_kernel(const _Float16* __restrict__ Q,
                 const _Float16* __restrict__ Km,
                 const _Float16* __restrict__ Vt,
                 _Float16* __restrict__ O)
{
    __shared__ alignas(64) _Float16 kt [2][32 * 256];   // [t_local][d]
    __shared__ alignas(64) _Float16 vtb[2][256 * 32];   // [d][t_local]
    __shared__ alignas(16) _Float16 lds_p[4][16 * 32];

    const int head = blockIdx.y;          // b*N + n
    const int b = head >> 2;
    const int n = head & 3;
    const int tid  = threadIdx.x;
    const int wave = tid >> 5;
    const int lane = tid & 31;
    const int lm = lane & 15;
    const int lh = lane >> 4;
    const int sb = blockIdx.x * 64 + wave * 16;

    // --- staging geometry (all 128 threads cooperate)
    const int krow = tid >> 2;            // 0..31  key row in tile
    const int kch  = (tid & 3) * 64;      // 64-half chunk within the 256-d row
    const _Float16* kgbase = Km + (size_t)(b * SS + krow) * HH + n * DD + kch;
    const unsigned klb[2] = {
        (unsigned)(size_t)&kt[0][krow * 256 + kch],
        (unsigned)(size_t)&kt[1][krow * 256 + kch] };

    const int vrow = tid * 2;             // 0..254  d row in tile (2 rows/thread)
    const _Float16* vgbase = Vt + (size_t)(head * DD + vrow) * SS;
    const unsigned vlb[2] = {
        (unsigned)(size_t)&vtb[0][vrow * 32],
        (unsigned)(size_t)&vtb[1][vrow * 32] };

    auto STAGEKV = [&](int buf, int tb) {
        // K tile: 8 x b128 per thread (128 bytes of one key row)
        unsigned long long kg = (unsigned long long)(size_t)(kgbase + (size_t)tb * HH);
        unsigned kl = klb[buf];
        #pragma unroll
        for (int u = 0; u < 4; ++u) {
            asm volatile("global_load_async_to_lds_b128 %0, %1, off"
                         :: "v"(kl + u * 32), "v"(kg + u * 32) : "memory");
            asm volatile("global_load_async_to_lds_b128 %0, %1, off offset:16"
                         :: "v"(kl + u * 32), "v"(kg + u * 32) : "memory");
        }
        // V tile: 2 d-rows x 64B per thread = 8 x b128
        #pragma unroll
        for (int r = 0; r < 2; ++r) {
            unsigned long long vg = (unsigned long long)(size_t)(vgbase + (size_t)r * SS + tb);
            unsigned vl = vlb[buf] + r * 64;
            asm volatile("global_load_async_to_lds_b128 %0, %1, off"
                         :: "v"(vl), "v"(vg) : "memory");
            asm volatile("global_load_async_to_lds_b128 %0, %1, off offset:16"
                         :: "v"(vl), "v"(vg) : "memory");
            asm volatile("global_load_async_to_lds_b128 %0, %1, off offset:32"
                         :: "v"(vl), "v"(vg) : "memory");
            asm volatile("global_load_async_to_lds_b128 %0, %1, off offset:48"
                         :: "v"(vl), "v"(vg) : "memory");
        }
    };

    // --- preload Q fragments (register resident, reused for all 64 key tiles)
    const _Float16* qrow = Q + (size_t)(b * SS + sb + lm) * HH + n * DD;
    v16h qa[8];
    #pragma unroll
    for (int dk = 0; dk < 8; ++dk) {
        v8h a0 = *(const v8h*)(qrow + dk * 32 + lh * 8);
        v8h a1 = *(const v8h*)(qrow + dk * 32 + 16 + lh * 8);
        #pragma unroll
        for (int i = 0; i < 8; ++i) { qa[dk][i] = a0[i]; qa[dk][8 + i] = a1[i]; }
    }

    float mrow[8], lrow[8];
    #pragma unroll
    for (int r = 0; r < 8; ++r) { mrow[r] = -1e30f; lrow[r] = 0.0f; }
    v8f o[16] = {};

    const float scale = 0.0625f;   // 1/sqrt(256)
    const int NT = SS / 32;        // 64 key tiles

    STAGEKV(0, 0);
    STAGEKV(1, 32);

    for (int it = 0; it < NT; ++it) {
        const int cur = it & 1;
        if (it + 1 < NT) asm volatile("s_wait_asynccnt 0x10" ::: "memory");
        else             asm volatile("s_wait_asynccnt 0x0"  ::: "memory");
        __syncthreads();                        // K/V tiles resident

        // ---- scores: S(16x32) = Q(16x256) * K^T(256x32), fragments from LDS
        v8f sc[2] = {};
        #pragma unroll
        for (int j = 0; j < 2; ++j) {
            #pragma unroll
            for (int dk = 0; dk < 8; ++dk) {
                const v8h* p = (const v8h*)&kt[cur][(j * 16 + lm) * 256 + dk * 32 + lh * 8];
                v8h t0 = p[0], t1 = p[2];
                v16h bf;
                #pragma unroll
                for (int i = 0; i < 8; ++i) { bf[i] = t0[i]; bf[8 + i] = t1[i]; }
                sc[j] = __builtin_amdgcn_wmma_f32_16x16x32_f16(false, qa[dk], false, bf, (short)0, sc[j], false, false);
            }
        }

        // ---- online softmax over the 32 keys
        #pragma unroll
        for (int r = 0; r < 8; ++r) {
            float v0 = sc[0][r] * scale;
            float v1 = sc[1][r] * scale;
            float mx = fmaxf(v0, v1);
            #pragma unroll
            for (int msk = 1; msk < 16; msk <<= 1) mx = fmaxf(mx, __shfl_xor(mx, msk, 32));
            float mnew = fmaxf(mrow[r], mx);
            float alpha = __expf(mrow[r] - mnew);
            float p0 = __expf(v0 - mnew);
            float p1 = __expf(v1 - mnew);
            float ps = p0 + p1;
            #pragma unroll
            for (int msk = 1; msk < 16; msk <<= 1) ps += __shfl_xor(ps, msk, 32);
            lrow[r] = lrow[r] * alpha + ps;
            mrow[r] = mnew;
            #pragma unroll
            for (int f = 0; f < 16; ++f) o[f][r] *= alpha;
            lds_p[wave][(lh * 8 + r) * 32 + lm]      = (_Float16)p0;
            lds_p[wave][(lh * 8 + r) * 32 + 16 + lm] = (_Float16)p1;
        }

        asm volatile("s_wait_dscnt 0x0" ::: "memory");

        // ---- P re-fragment (C-layout -> A-layout) via per-wave LDS tile
        v8h p0 = *(const v8h*)&lds_p[wave][lm * 32 + lh * 8];
        v8h p1 = *(const v8h*)&lds_p[wave][lm * 32 + 16 + lh * 8];
        v16h pa;
        #pragma unroll
        for (int i = 0; i < 8; ++i) { pa[i] = p0[i]; pa[8 + i] = p1[i]; }

        // ---- O(16x256) += P(16x32) * V(32x256), fragments from staged Vt tile
        #pragma unroll
        for (int f = 0; f < 16; ++f) {
            const v8h* p = (const v8h*)&vtb[cur][(f * 16 + lm) * 32 + lh * 8];
            v8h t0 = p[0], t1 = p[2];
            v16h bf;
            #pragma unroll
            for (int i = 0; i < 8; ++i) { bf[i] = t0[i]; bf[8 + i] = t1[i]; }
            o[f] = __builtin_amdgcn_wmma_f32_16x16x32_f16(false, pa, false, bf, (short)0, o[f], false, false);
        }

        asm volatile("s_wait_dscnt 0x0" ::: "memory");
        __syncthreads();                        // all waves done reading buf[cur]
        if (it + 2 < NT) STAGEKV(cur, (it + 2) * 32);
    }

    #pragma unroll
    for (int f = 0; f < 16; ++f) {
        const int d = f * 16 + lm;
        #pragma unroll
        for (int r = 0; r < 8; ++r) {
            const int s = sb + lh * 8 + r;
            O[(size_t)(b * SS + s) * HH + n * DD + d] = (_Float16)(o[f][r] / lrow[r]);
        }
    }
}

// ---------------------------------------------------------------------------
// Small fp32 support kernels
// ---------------------------------------------------------------------------
__global__ void embed_kernel(const int* __restrict__ ids, const float* __restrict__ emb,
                             float* __restrict__ X, _Float16* __restrict__ Xh)
{
    size_t i = (size_t)blockIdx.x * 256 + threadIdx.x;
    int h = (int)(i & (HH - 1));
    size_t tok = i >> 10;
    float v = emb[(size_t)ids[tok] * HH + h];
    X[i] = v; Xh[i] = (_Float16)v;
}

__global__ void ctx_kernel(const float* __restrict__ X, float* __restrict__ ctx)
{
    int b = blockIdx.x >> 2;
    int h = (blockIdx.x & 3) * 256 + threadIdx.x;
    const float* p = X + (size_t)b * SH + h;
    float s = 0.0f;
    for (int t = 0; t < SS; ++t) s += p[(size_t)t * HH];
    ctx[b * HH + h] = s * (1.0f / SS);
}

__global__ void dotrow_kernel(const float* __restrict__ ctx, const float* __restrict__ Wr,
                              const float* __restrict__ bias, float* __restrict__ outv, int R)
{
    __shared__ float red[8];
    int b = blockIdx.x / R, r = blockIdx.x % R;
    const float* c = ctx + b * HH;
    const float* w = Wr + (size_t)r * HH;
    int tid = threadIdx.x;
    float s = 0.0f;
    for (int i = tid; i < HH; i += 256) s += c[i] * w[i];
    for (int m = 16; m >= 1; m >>= 1) s += __shfl_xor(s, m, 32);
    if ((tid & 31) == 0) red[tid >> 5] = s;
    __syncthreads();
    if (tid == 0) {
        float S = 0.0f;
        for (int i = 0; i < 8; ++i) S += red[i];
        outv[blockIdx.x] = S + bias[r];
    }
}

__global__ void f32tof16_kernel(const float* __restrict__ src, _Float16* __restrict__ dst, int n)
{
    int i = blockIdx.x * 256 + threadIdx.x;
    if (i < n) dst[i] = (_Float16)src[i];
}

__global__ void gate_kernel(float* __restrict__ X, _Float16* __restrict__ Xh,
                            const float* __restrict__ Y, const float* __restrict__ dopnor)
{
    size_t i = (size_t)blockIdx.x * 256 + threadIdx.x;
    int b = (int)(i >> 21);
    float g = 1.0f / (1.0f + __expf(-Y[i]));
    float v = X[i] * (1.0f + dopnor[b * 2] * g);
    X[i] = v; Xh[i] = (_Float16)v;
}

__global__ void noise_kernel(float* __restrict__ X, _Float16* __restrict__ Xh,
                             const float* __restrict__ Y, const float* __restrict__ dopnor)
{
    size_t i = (size_t)blockIdx.x * 256 + threadIdx.x;
    int b = (int)(i >> 21);
    float v = X[i] + dopnor[b * 2 + 1] * tanhf(Y[i]);
    X[i] = v; Xh[i] = (_Float16)v;
}

__global__ void convq_kernel(const float* __restrict__ Y, _Float16* __restrict__ Qh,
                             const float* __restrict__ gain)
{
    size_t i = (size_t)blockIdx.x * 256 + threadIdx.x;
    int b = (int)(i >> 21);
    int n = (int)((i & (HH - 1)) >> 8);
    Qh[i] = (_Float16)(Y[i] * (1.0f + gain[b * NN + n]));
}

__global__ void conv_kernel(const float* __restrict__ Y, _Float16* __restrict__ Dh)
{
    size_t i = (size_t)blockIdx.x * 256 + threadIdx.x;
    Dh[i] = (_Float16)Y[i];
}

__global__ void transv_kernel(const float* __restrict__ Y, _Float16* __restrict__ Vt)
{
    size_t i = (size_t)blockIdx.x * 256 + threadIdx.x;   // over (b,n,d,s)
    int s = (int)(i & (SS - 1));
    size_t j = i >> 11;
    int d = (int)(j & (DD - 1));
    size_t k = j >> 8;
    int n = (int)(k & 3);
    int b = (int)(k >> 2);
    Vt[i] = (_Float16)Y[(size_t)(b * SS + s) * HH + n * DD + d];
}

__global__ void resadd_kernel(float* __restrict__ X, _Float16* __restrict__ Xh,
                              const float* __restrict__ Y)
{
    size_t i = (size_t)blockIdx.x * 256 + threadIdx.x;
    float v = X[i] + Y[i];
    X[i] = v; Xh[i] = (_Float16)v;
}

__global__ void silu_kernel(float* __restrict__ X, _Float16* __restrict__ Xh,
                            const float* __restrict__ Y)
{
    size_t i = (size_t)blockIdx.x * 256 + threadIdx.x;
    float y = Y[i];
    float v = y / (1.0f + __expf(-y));
    X[i] = v; Xh[i] = (_Float16)v;
}

__global__ __launch_bounds__(256)
void ln_kernel(float* __restrict__ X, _Float16* __restrict__ Xh,
               const float* __restrict__ g, const float* __restrict__ bta)
{
    __shared__ float red[2][8];
    const int row = blockIdx.x;
    const int tid = threadIdx.x;
    float* xr = X + (size_t)row * HH;
    float s = 0.0f, ss = 0.0f;
    #pragma unroll
    for (int i = 0; i < 4; ++i) { float v = xr[tid + i * 256]; s += v; ss += v * v; }
    for (int m = 16; m >= 1; m >>= 1) { s += __shfl_xor(s, m, 32); ss += __shfl_xor(ss, m, 32); }
    if ((tid & 31) == 0) { red[0][tid >> 5] = s; red[1][tid >> 5] = ss; }
    __syncthreads();
    float S = 0.0f, Q = 0.0f;
    #pragma unroll
    for (int i = 0; i < 8; ++i) { S += red[0][i]; Q += red[1][i]; }
    const float mu = S * (1.0f / HH);
    const float var = Q * (1.0f / HH) - mu * mu;
    const float inv = rsqrtf(var + 1e-5f);
    #pragma unroll
    for (int i = 0; i < 4; ++i) {
        int h = tid + i * 256;
        float v = (xr[h] - mu) * inv * g[h] + bta[h];
        xr[h] = v;
        Xh[(size_t)row * HH + h] = (_Float16)v;
    }
}

// ---------------------------------------------------------------------------
extern "C" void kernel_launch(void* const* d_in, const int* in_sizes, int n_in,
                              void* d_out, int out_size, void* d_ws, size_t ws_size,
                              hipStream_t stream)
{
    const int*   ids        = (const int*)  d_in[0];
    const float* embed      = (const float*)d_in[1];
    const float* nm_gate_w  = (const float*)d_in[2];
    const float* nm_gate_b  = (const float*)d_in[3];
    const float* nm_filter_w= (const float*)d_in[4];
    const float* nm_filter_b= (const float*)d_in[5];
    const float* nm_ctx_w   = (const float*)d_in[6];
    const float* nm_ctx_b   = (const float*)d_in[7];
    const float* q_w        = (const float*)d_in[8];
    const float* q_b        = (const float*)d_in[9];
    const float* k_w        = (const float*)d_in[10];
    const float* k_b        = (const float*)d_in[11];
    const float* v_w        = (const float*)d_in[12];
    const float* v_b        = (const float*)d_in[13];
    const float* gain_w     = (const float*)d_in[14];
    const float* gain_b     = (const float*)d_in[15];
    const float* attn_ow    = (const float*)d_in[16];
    const float* attn_ob    = (const float*)d_in[17];
    const float* en_w       = (const float*)d_in[18];
    const float* en_b       = (const float*)d_in[19];
    const float* plastic_w  = (const float*)d_in[20];
    const float* fix_w      = (const float*)d_in[21];
    const float* fix_b      = (const float*)d_in[22];
    const float* ln_g       = (const float*)d_in[23];
    const float* ln_b       = (const float*)d_in[24];
    const float* out_w      = (const float*)d_in[25];
    const float* out_b      = (const float*)d_in[26];

    char* ws = (char*)d_ws;
    size_t off = 0;
    auto take = [&](size_t bytes) -> char* {
        char* p = ws + off;
        off += (bytes + 255) & ~(size_t)255;
        return p;
    };
    float*    X32    = (float*)   take((size_t)BS * HH * 4);
    _Float16* X16    = (_Float16*)take((size_t)BS * HH * 2);
    float*    Y32    = (float*)   take((size_t)BS * HH * 4);
    _Float16* Q16    = (_Float16*)take((size_t)BS * HH * 2);
    _Float16* K16    = (_Float16*)take((size_t)BS * HH * 2);
    _Float16* Vt16   = (_Float16*)take((size_t)BS * HH * 2);
    _Float16* O16    = (_Float16*)take((size_t)BS * HH * 2);
    _Float16* W16    = (_Float16*)take((size_t)HH * HH * 2);
    float*    ctx32  = (float*)   take((size_t)BB * HH * 4);
    float*    dopnor = (float*)   take((size_t)BB * 2 * 4);
    float*    gain32 = (float*)   take((size_t)BB * NN * 4);

    const dim3 gHH(HH / 128, BS / 128);          // GEMM grid for Nout = H
    const int  WCONV = (HH * HH + 255) / 256;

    // ---- embed + neuromodulation
    embed_kernel<<<EW, 256, 0, stream>>>(ids, embed, X32, X16);
    ctx_kernel<<<16, 256, 0, stream>>>(X32, ctx32);
    dotrow_kernel<<<BB * 2, 256, 0, stream>>>(ctx32, nm_ctx_w, nm_ctx_b, dopnor, 2);

    f32tof16_kernel<<<WCONV, 256, 0, stream>>>(nm_gate_w, W16, HH * HH);
    wgemm_kernel<false, true><<<gHH, 256, 0, stream>>>(X16, W16, nm_gate_b, Y32, BS, HH, HH);
    gate_kernel<<<EW, 256, 0, stream>>>(X32, X16, Y32, dopnor);

    f32tof16_kernel<<<WCONV, 256, 0, stream>>>(nm_filter_w, W16, HH * HH);
    wgemm_kernel<false, true><<<gHH, 256, 0, stream>>>(X16, W16, nm_filter_b, Y32, BS, HH, HH);
    noise_kernel<<<EW, 256, 0, stream>>>(X32, X16, Y32, dopnor);

    // ---- layers
    for (int l = 0; l < LL; ++l) {
        const size_t wo = (size_t)l * HH * HH;
        ctx_kernel<<<16, 256, 0, stream>>>(X32, ctx32);
        dotrow_kernel<<<BB * NN, 256, 0, stream>>>(ctx32, gain_w + (size_t)l * NN * HH,
                                                   gain_b + l * NN, gain32, NN);
        // q
        f32tof16_kernel<<<WCONV, 256, 0, stream>>>(q_w + wo, W16, HH * HH);
        wgemm_kernel<false, true><<<gHH, 256, 0, stream>>>(X16, W16, q_b + l * HH, Y32, BS, HH, HH);
        convq_kernel<<<EW, 256, 0, stream>>>(Y32, Q16, gain32);
        // k
        f32tof16_kernel<<<WCONV, 256, 0, stream>>>(k_w + wo, W16, HH * HH);
        wgemm_kernel<false, true><<<gHH, 256, 0, stream>>>(X16, W16, k_b + l * HH, Y32, BS, HH, HH);
        conv_kernel<<<EW, 256, 0, stream>>>(Y32, K16);
        // v (transposed for attention B-fragments)
        f32tof16_kernel<<<WCONV, 256, 0, stream>>>(v_w + wo, W16, HH * HH);
        wgemm_kernel<false, true><<<gHH, 256, 0, stream>>>(X16, W16, v_b + l * HH, Y32, BS, HH, HH);
        transv_kernel<<<EW, 256, 0, stream>>>(Y32, Vt16);
        // attention
        attn_kernel<<<dim3(SS / 64, BB * NN), 128, 0, stream>>>(Q16, K16, Vt16, O16);
        // output proj + residual
        f32tof16_kernel<<<WCONV, 256, 0, stream>>>(attn_ow + wo, W16, HH * HH);
        wgemm_kernel<false, true><<<gHH, 256, 0, stream>>>(O16, W16, attn_ob + l * HH, Y32, BS, HH, HH);
        resadd_kernel<<<EW, 256, 0, stream>>>(X32, X16, Y32);
        // en + silu (replaces x)
        f32tof16_kernel<<<WCONV, 256, 0, stream>>>(en_w + wo, W16, HH * HH);
        wgemm_kernel<false, true><<<gHH, 256, 0, stream>>>(X16, W16, en_b + l * HH, Y32, BS, HH, HH);
        silu_kernel<<<EW, 256, 0, stream>>>(X32, X16, Y32);
        // fix + plastic, then residual
        f32tof16_kernel<<<WCONV, 256, 0, stream>>>(fix_w + wo, W16, HH * HH);
        wgemm_kernel<false, true><<<gHH, 256, 0, stream>>>(X16, W16, fix_b + l * HH, Y32, BS, HH, HH);
        f32tof16_kernel<<<WCONV, 256, 0, stream>>>(plastic_w + wo, W16, HH * HH);
        wgemm_kernel<true, false><<<gHH, 256, 0, stream>>>(X16, W16, nullptr, Y32, BS, HH, HH);
        resadd_kernel<<<EW, 256, 0, stream>>>(X32, X16, Y32);
        // layernorm
        ln_kernel<<<BS, 256, 0, stream>>>(X32, X16, ln_g + l * HH, ln_b + l * HH);
    }

    // ---- logits
    f32tof16_kernel<<<(VV * HH + 255) / 256, 256, 0, stream>>>(out_w, W16, VV * HH);
    wgemm_kernel<false, true><<<dim3(VV / 128, BS / 128), 256, 0, stream>>>(
        X16, W16, out_b, (float*)d_out, BS, HH, VV);
}
